// _GConv_45741401702776
// MI455X (gfx1250) — matrix operand order repaired and checked by reading, not compiled, not run on hardware
//
#include <hip/hip_runtime.h>
#include <hip/hip_bf16.h>
#include <math.h>
#include <stdint.h>

// GConv: y = GELU(causal_conv(x, multiscale_kernel) + x*skip) @ w_out^T + b_out
// x (16,256,8192) f32; kernels (9,1,256,32); skip (1,256); w_out (256,256); b_out (256)
//
// Conv = triangular Toeplitz matmul, M=batch(16), N=l-tile(16), K=t, bf16x3 split
// on V_WMMA_F32_16X16X32_BF16 with LDS-staged, double-buffered tiles.
// Output GEMM on V_WMMA_F32_16X16X4_F32.

typedef __attribute__((ext_vector_type(2)))  float  v2f;
typedef __attribute__((ext_vector_type(8)))  float  v8f;
typedef __attribute__((ext_vector_type(16))) __bf16 v16bf;

#define B_    16
#define D_    256
#define L_    8192
#define NS    9
#define KD    32
#define KPADF (L_ + 128)  // 128-zero front pad: uniform chunk count is branch-free

union FragBF { uint32_t u[8]; v16bf v; };

// round-to-nearest-even f32 -> bf16 (bits), and hi/lo split
__device__ __forceinline__ uint32_t bf16_rne(float f) {
  const uint32_t u = __float_as_uint(f);
  return (u + 0x7FFFu + ((u >> 16) & 1u)) >> 16;
}
__device__ __forceinline__ void bf16_split(float v, uint32_t& h, uint32_t& l) {
  h = bf16_rne(v);
  const float hf = __uint_as_float(h << 16);
  l = bf16_rne(v - hf);
}

// ---------------------------------------------------------------------------
// K1: build normalized multi-scale kernel rows (f32, 128-zero front pad).
// grid = (256), block = 256; block d owns its 8192 taps.
// ---------------------------------------------------------------------------
__global__ void gconv_build_kernel(const float* __restrict__ kernels,
                                   float* __restrict__ khat_pad) {
  const int d   = blockIdx.x;
  const int tid = threadIdx.x;

  float vals[32];
  float ss = 0.f;
#pragma unroll 4
  for (int n = 0; n < 32; ++n) {
    const int j = tid + (n << 8);
    int i, s0 = 0, len = 32;                 // pieces: 32,32,64,...,4096
    for (i = 0; i < NS; ++i) {
      len = 32 << ((i <= 1) ? 0 : (i - 1));
      if (j < s0 + len) break;
      s0 += len;
    }
    const int   local = j - s0;
    const int   s     = (i <= 1) ? 1 : (1 << (i - 1));
    float coord = ((float)local + 0.5f) / (float)s - 0.5f;
    coord = fminf(fmaxf(coord, 0.f), 31.f);
    const int   lo = (int)floorf(coord);
    const int   hi = min(lo + 1, 31);
    const float w  = coord - (float)lo;
    const float* kr = kernels + ((size_t)i * D_ + d) * KD;
    float v = kr[lo] * (1.f - w) + kr[hi] * w;
    v *= (float)(1 << (NS - 1 - i));         // multiplier 2^(NS-1-i)
    vals[n] = v;
    ss += v * v;
  }

  __shared__ float red[256];
  red[tid] = ss;
  __syncthreads();
  for (int off = 128; off > 0; off >>= 1) {
    if (tid < off) red[tid] += red[tid + off];
    __syncthreads();
  }
  const float inv = 1.0f / sqrtf(red[0]);

  float* row = khat_pad + (size_t)d * KPADF;
  if (tid < 128) row[tid] = 0.f;             // khat[p<0] == 0
#pragma unroll 4
  for (int n = 0; n < 32; ++n)
    row[128 + tid + (n << 8)] = vals[n] * inv;
}

// ---------------------------------------------------------------------------
// K1b: pack adjacent kernel pairs, split hi/lo bf16.
// kphl[d][q] = { pack(bf16h(row[q+1]), bf16h(row[q])), pack(lo,lo) }
// -> B-frag pair (khat[p], khat[p-1]) is one 8B entry at q = p+127.
// ---------------------------------------------------------------------------
__global__ void gconv_pack_k(const float* __restrict__ khat_pad,
                             uint2* __restrict__ kphl) {
  const int d = blockIdx.x;
  const float* row  = khat_pad + (size_t)d * KPADF;
  uint2*       orow = kphl     + (size_t)d * KPADF;
  for (int q = threadIdx.x; q < KPADF; q += blockDim.x) {
    const float v0 = row[min(q + 1, KPADF - 1)];   // -> [15:0]  (K even)
    const float v1 = row[q];                       // -> [31:16] (K odd)
    uint32_t h0, l0, h1, l1;
    bf16_split(v0, h0, l0);
    bf16_split(v1, h1, l1);
    orow[q] = make_uint2(h0 | (h1 << 16), l0 | (l1 << 16));
  }
}

// ---------------------------------------------------------------------------
// K2: transpose + split x into packed bf16 t-pairs:
// xphl[d][t/2][b] = { pack(bf16h(x[t]),bf16h(x[t+1])), pack(lo,lo) }
// -> A tile of a 32-tap chunk is a flat contiguous 2KB run.
// grid = (L/512, 256), block = 256.
// ---------------------------------------------------------------------------
__global__ void gconv_pack_x(const float* __restrict__ x,
                             uint2* __restrict__ xphl) {
  __shared__ float lds[16 * 512];
  const int d     = blockIdx.y;
  const int tbase = blockIdx.x * 512;
  const int tid   = threadIdx.x;

#pragma unroll
  for (int b = 0; b < 16; ++b) {
    const float2 v = *(const float2*)(x + ((size_t)b * D_ + d) * L_ + tbase + 2 * tid);
    *(float2*)(lds + b * 512 + 2 * tid) = v;
  }
  __syncthreads();

  uint2* ob = xphl + ((size_t)d * (L_ / 2) + (tbase >> 1)) * 16;
#pragma unroll
  for (int i = 0; i < 16; ++i) {
    const int o   = tid + i * 256;      // o = t2off*16 + b
    const int b   = o & 15;
    const int t2o = o >> 4;
    const float v0 = lds[b * 512 + 2 * t2o];
    const float v1 = lds[b * 512 + 2 * t2o + 1];
    uint32_t h0, l0, h1, l1;
    bf16_split(v0, h0, l0);
    bf16_split(v1, h1, l1);
    ob[o] = make_uint2(h0 | (h1 << 16), l0 | (l1 << 16));
  }
}

// ---------------------------------------------------------------------------
// K3: causal Toeplitz conv, bf16x3, LDS-staged double-buffered tiles.
// Block = 8 waves sharing channel d, l-range 128 (wave w -> l0 = base+16w).
// Per 32-tap chunk: one 2KB A tile + one 1.3KB B window staged once for all
// 8 waves; fragments via ds_load_b64; 3 WMMA/wave/chunk.
// grid = (L/128, 256), block = 256.
// ---------------------------------------------------------------------------
__global__ void gconv_conv_wmma(const uint2* __restrict__ xphl,
                                const uint2* __restrict__ kphl,
                                const float* __restrict__ x,
                                const float* __restrict__ skip_w,
                                float* __restrict__ g) {
  __shared__ uint2 lda[2][16 * 17];     // [khalf][b], stride 17: bank-safe
  __shared__ uint2 ldb[2][160];         // shared kernel-pair window

  const int d       = blockIdx.y;
  const int tid     = threadIdx.x;
  const int lane    = tid & 31;
  const int wave    = tid >> 5;
  const int l0_base = blockIdx.x * 128;
  const int half    = lane >> 4;        // K-octet select (ISA 16-bit A/B layout)
  const int nl      = lane & 15;        // A: M/batch; B,C,D: N/l

  const uint2* xrow = xphl + (size_t)d * (L_ / 2) * 16;
  const uint2* krow = kphl + (size_t)d * KPADF;

  const int NCH = (l0_base >> 5) + 4;   // uniform chunks; pad zeroes surplus
  const int t2o = tid >> 4, bb = tid & 15;

  // preload chunk 0
  uint2 areg = xrow[tid];
  uint2 breg;
  if (tid < 160) breg = krow[min(l0_base + 97 + tid, KPADF - 1)];
  lda[0][t2o * 17 + bb] = areg;
  if (tid < 160) ldb[0][tid] = breg;
  __syncthreads();

  v8f acc = {};
  for (int c = 0; c < NCH; ++c) {
    const int cur = c & 1, nxt = cur ^ 1;

    if (c + 1 < NCH) {                  // global loads for next chunk
      areg = xrow[(c + 1) * 256 + tid];
      const int qbn = l0_base - ((c + 1) << 5) + 97;
      if (tid < 160) breg = krow[min(qbn + tid, KPADF - 1)];
    }
    if (c + 2 < NCH)                    // global_prefetch_b8 two chunks ahead
      __builtin_prefetch((const void*)&xrow[(c + 2) * 256 + tid], 0, 1);

    // fragments from LDS
    FragBF ah, al, bh, bl;
#pragma unroll
    for (int j = 0; j < 8; ++j) {
      const int khalf = (j < 4) ? (half * 4 + j) : (8 + half * 4 + (j - 4));
      const uint2 av = lda[cur][khalf * 17 + nl];
      ah.u[j] = av.x;
      al.u[j] = av.y;
      // B[k][n] = khat[l0-t0+n-k]; window-local index, zeros handle causality
      const uint2 bv = ldb[cur][wave * 16 + nl + 30 - 2 * khalf];
      bh.u[j] = bv.x;
      bl.u[j] = bv.y;
    }

    // bf16x3: hi*hi + hi*lo + lo*hi  (~fp32 accuracy, bf16 pipe rate)
    acc = __builtin_amdgcn_wmma_f32_16x16x32_bf16(false, ah.v, false, bh.v,
                                                  (short)0, acc, false, false);
    acc = __builtin_amdgcn_wmma_f32_16x16x32_bf16(false, ah.v, false, bl.v,
                                                  (short)0, acc, false, false);
    acc = __builtin_amdgcn_wmma_f32_16x16x32_bf16(false, al.v, false, bh.v,
                                                  (short)0, acc, false, false);

    if (c + 1 < NCH) {                  // fill other buffer, one barrier/chunk
      lda[nxt][t2o * 17 + bb] = areg;
      if (tid < 160) ldb[nxt][tid] = breg;
      __syncthreads();
    }
  }

  // Epilogue: + x*skip, exact GELU, write g (B,D,L)
  const float sw = skip_w[d];
  const int   l0 = l0_base + wave * 16;
  const int   l  = l0 + nl;
#pragma unroll
  for (int r = 0; r < 8; ++r) {
    const int   bi = r + half * 8;      // C/D: VGPR r -> M=r (+8 upper half)
    const float xv = x[((size_t)bi * D_ + d) * L_ + l];
    float v = acc[r] + xv * sw;
    v = 0.5f * v * (1.0f + erff(v * 0.70710678118654752440f));
    g[((size_t)bi * D_ + d) * L_ + l] = v;
  }
}

// ---------------------------------------------------------------------------
// K4: out[b,m,l] = sum_k w_out[m,k]*g[b,k,l] + b_out[m]  (f32 WMMA, K=256)
// w_out M-tile in LDS (stride 260: conflict-free).  grid = (L/128, 16, 16).
// ---------------------------------------------------------------------------
#define LDW 260
__global__ void gconv_out_gemm(const float* __restrict__ g,
                               const float* __restrict__ w_out,
                               const float* __restrict__ b_out,
                               float* __restrict__ out) {
  __shared__ float ldsw[16 * LDW];
  const int bidx = blockIdx.z;
  const int m0   = blockIdx.y * 16;
  const int tid  = threadIdx.x;

  for (int i = tid; i < 16 * 256; i += 256) {
    const int m = i >> 8, k = i & 255;
    ldsw[m * LDW + k] = w_out[(size_t)(m0 + m) * D_ + k];
  }
  __syncthreads();

  const int lane = tid & 31, wave = tid >> 5;
  const int half = lane >> 4, nl = lane & 15;
  const int l0   = (blockIdx.x * 8 + wave) * 16;
  const float* gb = g + (size_t)bidx * D_ * L_;

  v8f acc = {};
#pragma unroll 4
  for (int k0 = 0; k0 < 256; k0 += 4) {
    const int kp = half * 2;
    const float2 av = *(const float2*)(&ldsw[nl * LDW + k0 + kp]);  // A[m][k]
    v2f a; a.x = av.x; a.y = av.y;
    v2f b;
    b.x = gb[(size_t)(k0 + kp) * L_ + l0 + nl];                     // B[k][n]
    b.y = gb[(size_t)(k0 + kp + 1) * L_ + l0 + nl];
    acc = __builtin_amdgcn_wmma_f32_16x16x4_f32(
        false, a, false, b, (short)0, acc, false, false);
  }

  const int l = l0 + nl;
#pragma unroll
  for (int r = 0; r < 8; ++r) {
    const int m = m0 + r + half * 8;
    out[((size_t)bidx * D_ + m) * L_ + l] = acc[r] + b_out[m];
  }
}

// ---------------------------------------------------------------------------
extern "C" void kernel_launch(void* const* d_in, const int* in_sizes, int n_in,
                              void* d_out, int out_size, void* d_ws, size_t ws_size,
                              hipStream_t stream) {
  const float* x       = (const float*)d_in[0];
  const float* kernels = (const float*)d_in[1];
  const float* skip_w  = (const float*)d_in[2];
  const float* w_out   = (const float*)d_in[3];
  const float* b_out   = (const float*)d_in[4];
  float*       out     = (float*)d_out;

  // workspace layout (all 8B-aligned):
  float* khat_pad = (float*)d_ws;                               // 256*8320 f32  (8.5 MB)
  uint2* kphl     = (uint2*)(khat_pad + (size_t)D_ * KPADF);    // 256*8320 u2  (17 MB)
  uint2* xphl     = kphl + (size_t)D_ * KPADF;                  // 256*4096*16 u2 (134 MB)
  float* g        = (float*)(xphl + (size_t)D_ * (L_ / 2) * 16);// 16*256*8192 f32 (134 MB)

  gconv_build_kernel<<<dim3(D_),                dim3(256), 0, stream>>>(kernels, khat_pad);
  gconv_pack_k      <<<dim3(D_),                dim3(256), 0, stream>>>(khat_pad, kphl);
  gconv_pack_x      <<<dim3(L_ / 512, D_),      dim3(256), 0, stream>>>(x, xphl);
  gconv_conv_wmma   <<<dim3(L_ / 128, D_),      dim3(256), 0, stream>>>(xphl, kphl, x, skip_w, g);
  gconv_out_gemm    <<<dim3(L_ / 128, D_ / 16, B_), dim3(256), 0, stream>>>(g, w_out, b_out, out);
}
